// get_model_87660282511912
// MI455X (gfx1250) — matrix operand order repaired and checked by reading, not compiled, not run on hardware
//
#include <hip/hip_runtime.h>
#include <cstdint>
#include <cstddef>

#define KB 8
#define KN 4096

typedef __attribute__((ext_vector_type(8)))  unsigned short us8;
typedef __attribute__((ext_vector_type(16))) unsigned short us16;
typedef __attribute__((ext_vector_type(16))) __bf16         v16bf;
typedef __attribute__((ext_vector_type(8)))  float          v8f;

__device__ __forceinline__ unsigned short f2bf(float f) {
    unsigned u = __float_as_uint(f);
    u += 0x7FFFu + ((u >> 16) & 1u);          // round-to-nearest-even
    return (unsigned short)(u >> 16);
}
__device__ __forceinline__ float bf2f(unsigned short h) {
    return __uint_as_float(((unsigned)h) << 16);
}

// CDNA5 async global->LDS copy, 16B per lane. Tracked by ASYNCcnt.
__device__ __forceinline__ void async_b128(unsigned lds_addr, const void* gaddr) {
    asm volatile("global_load_async_to_lds_b128 %0, %1, off"
                 :: "v"(lds_addr), "v"((unsigned long long)(size_t)gaddr)
                 : "memory");
}

// ---------------- layout / prep kernels ----------------

__global__ void k_transpose_xyz(const float* __restrict__ in, float* __restrict__ out) {
    int t = blockIdx.x * blockDim.x + threadIdx.x;
    if (t >= KB * KN * 3) return;
    int c = t % 3, n = (t / 3) % KN, b = t / (3 * KN);
    out[t] = in[(size_t)b * 3 * KN + (size_t)c * KN + n];
}

__global__ void k_l0cat(const float* __restrict__ xyzp, float* __restrict__ out) {
    int t = blockIdx.x * blockDim.x + threadIdx.x;
    if (t >= KB * KN * 6) return;
    int c = t % 6; size_t pn = (size_t)t / 6;
    out[t] = xyzp[pn * 3 + (c % 3)];
}

__global__ void k_prep_weight(const float* __restrict__ W, unsigned short* __restrict__ Wb,
                              int Cout, int K, int Cp, int Kp) {
    int i = blockIdx.x * blockDim.x + threadIdx.x;
    if (i >= Cp * Kp) return;
    int r = i / Kp, c = i % Kp;
    float v = (r < Cout && c < K) ? W[(size_t)r * K + c] : 0.f;
    Wb[i] = f2bf(v);
}

// ---------------- geometry kernels ----------------

__global__ __launch_bounds__(256)
void k_fps(const float* __restrict__ xyz, int n, int npoint, int* __restrict__ out) {
    __shared__ float sdist[4096];
    __shared__ float sval[256];
    __shared__ int   sidx[256];
    __shared__ int   sfar;
    int b = blockIdx.x;
    const float* p = xyz + (size_t)b * n * 3;
    for (int i = threadIdx.x; i < n; i += 256) sdist[i] = 1e10f;
    if (threadIdx.x == 0) sfar = 0;
    __syncthreads();
    for (int it = 0; it < npoint; ++it) {
        int far = sfar;
        if (threadIdx.x == 0) out[b * npoint + it] = far;
        float cx = p[far * 3 + 0], cy = p[far * 3 + 1], cz = p[far * 3 + 2];
        float best = -1.f; int bi = 0;
        for (int i = threadIdx.x; i < n; i += 256) {
            float dx = p[i * 3] - cx, dy = p[i * 3 + 1] - cy, dz = p[i * 3 + 2] - cz;
            float d = dx * dx + dy * dy + dz * dz;
            float dm = fminf(sdist[i], d); sdist[i] = dm;
            if (dm > best) { best = dm; bi = i; }
        }
        sval[threadIdx.x] = best; sidx[threadIdx.x] = bi;
        __syncthreads();
        for (int s = 128; s > 0; s >>= 1) {
            if (threadIdx.x < s && sval[threadIdx.x + s] > sval[threadIdx.x]) {
                sval[threadIdx.x] = sval[threadIdx.x + s];
                sidx[threadIdx.x] = sidx[threadIdx.x + s];
            }
            __syncthreads();
        }
        if (threadIdx.x == 0) sfar = sidx[0];
        __syncthreads();
    }
}

__global__ void k_gather(const float* __restrict__ pts, const int* __restrict__ idx,
                         float* __restrict__ out, int n, int s, int C) {
    int t = blockIdx.x * blockDim.x + threadIdx.x;
    if (t >= KB * s * C) return;
    int c = t % C, q = (t / C) % s, b = t / (C * s);
    out[t] = pts[((size_t)b * n + idx[b * s + q]) * C + c];
}

__global__ void k_ballquery(const float* __restrict__ xyz, const float* __restrict__ nxyz,
                            int n, int s, int ns, float r2, int* __restrict__ gidx) {
    int t = blockIdx.x * blockDim.x + threadIdx.x;
    if (t >= KB * s) return;
    int b = t / s;
    const float* p = xyz + (size_t)b * n * 3;
    float cx = nxyz[t * 3], cy = nxyz[t * 3 + 1], cz = nxyz[t * 3 + 2];
    int* o = gidx + (size_t)t * ns;
    int cnt = 0, first = 0; bool has = false;
    for (int i = 0; i < n && cnt < ns; ++i) {
        float dx = p[i * 3] - cx, dy = p[i * 3 + 1] - cy, dz = p[i * 3 + 2] - cz;
        float d = dx * dx + dy * dy + dz * dz;
        if (d <= r2) { if (!has) { first = i; has = true; } o[cnt++] = i; }
    }
    for (; cnt < ns; ++cnt) o[cnt] = first;
}

// grouped = concat(gathered_xyz - new_xyz, gathered_points), bf16, K-padded
__global__ void k_group(const float* __restrict__ xyz, const float* __restrict__ nxyz,
                        const int* __restrict__ gidx, const float* __restrict__ pts,
                        int n, int s, int ns, int Cp, int Kpad, unsigned short* __restrict__ out) {
    size_t t = (size_t)blockIdx.x * blockDim.x + threadIdx.x;
    size_t total = (size_t)KB * s * ns * Kpad;
    if (t >= total) return;
    int c = (int)(t % Kpad);
    size_t row = t / Kpad;
    int k = (int)(row % ns);
    size_t bq = row / ns;                 // b*s+q
    int b = (int)(bq / s);
    int gi = gidx[bq * ns + k];
    float v = 0.f;
    if (c < 3)              v = xyz[((size_t)b * n + gi) * 3 + c] - nxyz[bq * 3 + c];
    else if (c < 3 + Cp)    v = pts[((size_t)b * n + gi) * Cp + (c - 3)];
    out[t] = f2bf(v);
}

__global__ void k_groupall(const float* __restrict__ l2x, const float* __restrict__ l2p,
                           int s, int Cp, int Kpad, unsigned short* __restrict__ out) {
    size_t t = (size_t)blockIdx.x * blockDim.x + threadIdx.x;
    size_t total = (size_t)KB * s * Kpad;
    if (t >= total) return;
    int c = (int)(t % Kpad);
    size_t row = t / Kpad;                // b*s+k
    float v = 0.f;
    if (c < 3)            v = l2x[row * 3 + c];
    else if (c < 3 + Cp)  v = l2p[row * Cp + (c - 3)];
    out[t] = f2bf(v);
}

__global__ void k_maxpool(const unsigned short* __restrict__ in, float* __restrict__ out,
                          int PS, int ns, int C, int ld) {
    int t = blockIdx.x * blockDim.x + threadIdx.x;
    if (t >= PS * C) return;
    int c = t % C, ps = t / C;
    const unsigned short* base = in + (size_t)ps * ns * ld + c;
    float m = -3.0e38f;
    for (int k = 0; k < ns; ++k) m = fmaxf(m, bf2f(base[(size_t)k * ld]));
    out[t] = m;
}

// out[row][0:C1) = A[row], out[row][C1:C1+C2) = B[row or row/n1], bf16, K-padded
__global__ void k_concat2(const float* __restrict__ A, int C1, const float* __restrict__ Bp, int C2,
                          int n1, int bcast, int Kpad, unsigned short* __restrict__ out, size_t rows) {
    size_t t = (size_t)blockIdx.x * blockDim.x + threadIdx.x;
    size_t total = rows * (size_t)Kpad;
    if (t >= total) return;
    int c = (int)(t % Kpad);
    size_t row = t / Kpad;
    float v = 0.f;
    if (c < C1) v = A[row * C1 + c];
    else if (c < C1 + C2) {
        size_t r2 = bcast ? (row / n1) : row;
        v = Bp[r2 * C2 + (c - C1)];
    }
    out[t] = f2bf(v);
}

__global__ void k_interp3(const float* __restrict__ xyz1, const float* __restrict__ xyz2,
                          const float* __restrict__ pts2, float* __restrict__ out,
                          int n1, int n2, int C) {
    int t = blockIdx.x * blockDim.x + threadIdx.x;
    if (t >= KB * n1) return;
    int b = t / n1;
    const float* p1 = &xyz1[(size_t)t * 3];
    const float* p2 = &xyz2[(size_t)b * n2 * 3];
    float x = p1[0], y = p1[1], z = p1[2];
    float d0 = 3e38f, d1 = 3e38f, d2 = 3e38f; int i0 = 0, i1 = 0, i2 = 0;
    for (int j = 0; j < n2; ++j) {
        float dx = x - p2[j * 3], dy = y - p2[j * 3 + 1], dz = z - p2[j * 3 + 2];
        float d = dx * dx + dy * dy + dz * dz;
        if (d < d0)      { d2 = d1; i2 = i1; d1 = d0; i1 = i0; d0 = d; i0 = j; }
        else if (d < d1) { d2 = d1; i2 = i1; d1 = d; i1 = j; }
        else if (d < d2) { d2 = d; i2 = j; }
    }
    float w0 = 1.f / (d0 + 1e-8f), w1 = 1.f / (d1 + 1e-8f), w2 = 1.f / (d2 + 1e-8f);
    float ws = w0 + w1 + w2; w0 /= ws; w1 /= ws; w2 /= ws;
    const float* f0 = &pts2[((size_t)b * n2 + i0) * C];
    const float* f1 = &pts2[((size_t)b * n2 + i1) * C];
    const float* f2 = &pts2[((size_t)b * n2 + i2) * C];
    float* o = &out[(size_t)t * C];
    for (int c = 0; c < C; ++c) o[c] = w0 * f0[c] + w1 * f1[c] + w2 * f2[c];
}

// ---------------- WMMA GEMM (bf16 -> f32 acc), fused bias/BN/ReLU ----------------
// Block: 256 threads = 8 wave32 (4x2 grid), tile M128 x N64 x K64.
// Double-buffered LDS filled by global_load_async_to_lds_b128; ASYNCcnt pipelining.
// Each wave: 2x2 accumulators x 2 k-steps = 8 v_wmma_f32_16x16x32_bf16 per K-tile.

alignas(128) __device__ __constant__ int k_dummy_pad;  // keep alignment attrs used

__device__ __forceinline__ void stage_tiles_async(
    const unsigned short* __restrict__ X, const unsigned short* __restrict__ Wg,
    size_t rowBase, int colBase, int Kp, int k0,
    unsigned short (*Xd)[72], unsigned short (*Wd)[80], int tid) {
#pragma unroll
    for (int i = 0; i < 4; ++i) {                 // X tile: 128 rows x 8 chunks = 1024
        int c = tid + 256 * i;
        int r = c >> 3, cc = (c & 7) * 8;
        async_b128((unsigned)(size_t)&Xd[r][cc],
                   &X[(rowBase + r) * (size_t)Kp + k0 + cc]);
    }
#pragma unroll
    for (int i = 0; i < 2; ++i) {                 // W tile: 64 rows x 8 chunks = 512
        int c = tid + 256 * i;
        int r = c >> 3, cc = (c & 7) * 8;
        async_b128((unsigned)(size_t)&Wd[r][cc],
                   &Wg[(size_t)(colBase + r) * Kp + k0 + cc]);
    }
}

__global__ __launch_bounds__(256)
void k_wmma_gemm(const unsigned short* __restrict__ X,   // [M][Kp] bf16 bits
                 const unsigned short* __restrict__ W,   // [Cp][Kp] bf16 bits
                 const float* __restrict__ bias,
                 const float* __restrict__ gam,
                 const float* __restrict__ bet,
                 unsigned short* __restrict__ Ybf,       // [M][ldOut] (+colOff)
                 float* __restrict__ Yf,                 // optional [M][Cout]
                 int M, int Kp, int Cout, int ldOut, int colOff, int mode) {
    alignas(128) __shared__ unsigned short Xs[2][128][72];  // 64 + pad (rows 144B apart)
    alignas(128) __shared__ unsigned short Ws2[2][64][80];  // 64 + pad (rows 160B, 32B-aligned)
    const int tid  = threadIdx.x;
    const int wave = tid >> 5, lane = tid & 31;
    const int wm = wave & 3, wn = wave >> 2;
    const int hl = lane >> 4, l15 = lane & 15;
    const size_t rowBase = (size_t)blockIdx.x * 128;
    const int    colBase = blockIdx.y * 64;

    v8f acc[2][2] = {};

    stage_tiles_async(X, W, rowBase, colBase, Kp, 0, Xs[0], Ws2[0], tid);

    const int ntk = Kp >> 6;
    for (int kt = 0; kt < ntk; ++kt) {
        const int cur = kt & 1;
        if (kt + 1 < ntk) {
            stage_tiles_async(X, W, rowBase, colBase, Kp, (kt + 1) << 6,
                              Xs[cur ^ 1], Ws2[cur ^ 1], tid);
            // 6 async ops just issued for the next tile; previous tile's 6 retire in order.
            asm volatile("s_wait_asynccnt 0x6" ::: "memory");
        } else {
            asm volatile("s_wait_asynccnt 0x0" ::: "memory");
        }
        __syncthreads();

        const unsigned short (*Xc)[72] = Xs[cur];
        const unsigned short (*Wc)[80] = Ws2[cur];
#pragma unroll
        for (int kk = 0; kk < 64; kk += 32) {
            // A fragments: lane l15 = row; K = {kk+8h..+7} U {kk+16+8h..+7}
            v16bf afr[2], bfr[2];
#pragma unroll
            for (int mt = 0; mt < 2; ++mt) {
                int r = wm * 32 + mt * 16 + l15;
                us8 lo = *reinterpret_cast<const us8*>(&Xc[r][kk + hl * 8]);
                us8 hi = *reinterpret_cast<const us8*>(&Xc[r][kk + 16 + hl * 8]);
                union { us16 v; us8 h[2]; } u;
                u.h[0] = lo; u.h[1] = hi;
                afr[mt] = __builtin_bit_cast(v16bf, u.v);
            }
            // B fragments: lane l15 = column; K = {kk+8h..+7} U {kk+16+8h..+7}
#pragma unroll
            for (int nt = 0; nt < 2; ++nt) {
                int r = wn * 32 + nt * 16 + l15;
                union { us16 v; us8 h[2]; } ub;
                ub.h[0] = *reinterpret_cast<const us8*>(&Wc[r][kk + hl * 8]);
                ub.h[1] = *reinterpret_cast<const us8*>(&Wc[r][kk + 16 + hl * 8]);
                bfr[nt] = __builtin_bit_cast(v16bf, ub.v);
            }
#pragma unroll
            for (int mt = 0; mt < 2; ++mt)
#pragma unroll
                for (int nt = 0; nt < 2; ++nt)
                    acc[mt][nt] = __builtin_amdgcn_wmma_f32_16x16x32_bf16(
                        false, afr[mt], false, bfr[nt], (short)0, acc[mt][nt], false, false);
        }
        __syncthreads();
    }

    const float inv = 0.9999950000374997f;   // 1/sqrt(1+1e-5)
#pragma unroll
    for (int mt = 0; mt < 2; ++mt) {
#pragma unroll
        for (int nt = 0; nt < 2; ++nt) {
            int n = colBase + wn * 32 + nt * 16 + l15;
            if (n >= Cout) continue;
            float bv = bias ? bias[n] : 0.f;
            float gv = 0.f, bev = 0.f;
            if (mode == 1) { gv = gam[n]; bev = bet[n]; }
#pragma unroll
            for (int i = 0; i < 8; ++i) {
                // C/D layout: VGPR i -> M = i + 8*(lane>=16), N = lane&15
                size_t m = rowBase + wm * 32 + mt * 16 + i + 8 * hl;
                float y = acc[mt][nt][i] + bv;
                if (mode == 1) y = fmaxf(gv * (y * inv) + bev, 0.f);
                if (Ybf) Ybf[m * ldOut + colOff + n] = f2bf(y);
                if (Yf)  Yf[m * Cout + n] = y;
            }
        }
    }
}

// head tail: 128 -> 32 -> 1 -> 1, all plain linear
__global__ __launch_bounds__(256)
void k_head_tail(const unsigned short* __restrict__ f128,
                 const float* __restrict__ W4, const float* __restrict__ b4,
                 const float* __restrict__ W5, const float* __restrict__ b5,
                 const float* __restrict__ W6, const float* __restrict__ b6,
                 float* __restrict__ out, int M) {
    int t = blockIdx.x * blockDim.x + threadIdx.x;
    if (t >= M) return;
    float x[128];
    const unsigned short* p = f128 + (size_t)t * 128;
    for (int k = 0; k < 128; ++k) x[k] = bf2f(p[k]);
    float t1 = b5[0];
    for (int o = 0; o < 32; ++o) {
        float s = b4[o];
        const float* w = W4 + o * 128;
        for (int k = 0; k < 128; ++k) s += w[k] * x[k];
        t1 += W5[o] * s;
    }
    out[t] = W6[0] * t1 + b6[0];
}

// ---------------- host orchestration ----------------

extern "C" void kernel_launch(void* const* d_in, const int* in_sizes, int n_in,
                              void* d_out, int out_size, void* d_ws, size_t ws_size,
                              hipStream_t stream) {
    (void)in_sizes; (void)n_in; (void)out_size; (void)ws_size;
    char* wsb = (char*)d_ws; size_t off = 0;
    auto alloc = [&](size_t bytes) -> void* {
        off = (off + 255) & ~(size_t)255;
        void* p = wsb + off; off += bytes; return p;
    };
    auto ru = [](int x, int m) { return (x + m - 1) / m * m; };

    const float* xyz_in = (const float*)d_in[0];

    float* xyzp  = (float*)alloc((size_t)KB * KN * 3 * 4);
    float* l0cat = (float*)alloc((size_t)KB * KN * 6 * 4);
    int*   fidx  = (int*)alloc((size_t)KB * 512 * 4);
    int*   gidx  = (int*)alloc((size_t)KB * 512 * 64 * 4);
    float* l1x   = (float*)alloc((size_t)KB * 512 * 3 * 4);
    float* l2x   = (float*)alloc((size_t)KB * 128 * 3 * 4);
    float* l1p   = (float*)alloc((size_t)KB * 512 * 128 * 4);
    float* l2p   = (float*)alloc((size_t)KB * 128 * 256 * 4);
    float* l3p   = (float*)alloc((size_t)KB * 1024 * 4);
    float* l2pn  = (float*)alloc((size_t)KB * 128 * 256 * 4);
    float* l1pn  = (float*)alloc((size_t)KB * 512 * 128 * 4);
    float* interp = (float*)alloc((size_t)KB * KN * 128 * 4);
    unsigned short* l0pb[2];
    l0pb[0] = (unsigned short*)alloc((size_t)KB * KN * 128 * 2);
    l0pb[1] = (unsigned short*)alloc((size_t)KB * KN * 128 * 2);
    unsigned short* ping = (unsigned short*)alloc((size_t)262144 * 128 * 2);
    unsigned short* pong = (unsigned short*)alloc((size_t)262144 * 128 * 2);

    auto g1 = [](size_t total) { return dim3((unsigned)((total + 255) / 256)); };

    k_transpose_xyz<<<g1(KB * KN * 3), dim3(256), 0, stream>>>(xyz_in, xyzp);
    k_l0cat<<<g1(KB * KN * 6), dim3(256), 0, stream>>>(xyzp, l0cat);

    auto prepw = [&](const float* W, int C, int K) -> unsigned short* {
        int Kp = ru(K, 64), Cp = ru(C, 64);
        unsigned short* Wb = (unsigned short*)alloc((size_t)Kp * Cp * 2);
        k_prep_weight<<<g1((size_t)Kp * Cp), dim3(256), 0, stream>>>(W, Wb, C, K, Cp, Kp);
        return Wb;
    };
    auto gemm = [&](const unsigned short* Xb, int M, int Kp, const unsigned short* Wb,
                    int C, int Cp, const float* bias, const float* g, const float* be,
                    int mode, unsigned short* Ybf, float* Yf, int ldOut, int colOff) {
        dim3 grid(M / 128, Cp / 64);
        k_wmma_gemm<<<grid, dim3(256), 0, stream>>>(Xb, Wb, bias, g, be, Ybf, Yf,
                                                    M, Kp, C, ldOut, colOff, mode);
    };
    auto run_mlp = [&](unsigned short* in, int M, int cin, const int* chans, int nl, int base,
                       float* finalF32, unsigned short* finalBf, int finalLd) -> unsigned short* {
        int K = cin;
        for (int j = 0; j < nl; ++j) {
            const float* W  = (const float*)d_in[base + 4 * j + 0];
            const float* bb = (const float*)d_in[base + 4 * j + 1];
            const float* gg = (const float*)d_in[base + 4 * j + 2];
            const float* be = (const float*)d_in[base + 4 * j + 3];
            int C = chans[j], Kp = ru(K, 64), Cp = ru(C, 64);
            unsigned short* Wb = prepw(W, C, K);
            bool last = (j == nl - 1);
            unsigned short* ob = (last && finalBf) ? finalBf : ((in == ping) ? pong : ping);
            int ld = (last && finalBf) ? finalLd : Cp;
            gemm(in, M, Kp, Wb, C, Cp, bb, gg, be, 1, ob, last ? finalF32 : nullptr, ld, 0);
            in = ob; K = C;
        }
        return in;
    };

    const int sa_ch[3][3] = {{64, 64, 128}, {128, 128, 256}, {256, 512, 1024}};
    const int fp_ch[3][3] = {{256, 256, 0}, {256, 128, 0}, {128, 128, 128}};
    const int fp_cum[6]   = {0, 2, 4, 7, 9, 11};
    const float radii[2][2] = {{0.2f, 0.4f}, {0.3f, 0.5f}};

    for (int br = 0; br < 2; ++br) {
        int saB0 = 1 + 12 * (3 * br + 0), saB1 = 1 + 12 * (3 * br + 1), saB2 = 1 + 12 * (3 * br + 2);
        int fpB0 = 73 + 4 * fp_cum[3 * br + 0], fpB1 = 73 + 4 * fp_cum[3 * br + 1],
            fpB2 = 73 + 4 * fp_cum[3 * br + 2];
        float r1 = radii[br][0], r2 = radii[br][1];

        // ---- SA1: npoint=512, ns=64, 6 -> 64,64,128 (Kpad 64) ----
        k_fps<<<dim3(KB), dim3(256), 0, stream>>>(xyzp, KN, 512, fidx);
        k_gather<<<g1(KB * 512 * 3), dim3(256), 0, stream>>>(xyzp, fidx, l1x, KN, 512, 3);
        k_ballquery<<<g1(KB * 512), dim3(256), 0, stream>>>(xyzp, l1x, KN, 512, 64, r1 * r1, gidx);
        k_group<<<g1((size_t)KB * 512 * 64 * 64), dim3(256), 0, stream>>>(
            xyzp, l1x, gidx, xyzp, KN, 512, 64, 3, 64, ping);
        unsigned short* o1 = run_mlp(ping, KB * 512 * 64, 6, sa_ch[0], 3, saB0, nullptr, nullptr, 0);
        k_maxpool<<<g1(KB * 512 * 128), dim3(256), 0, stream>>>(o1, l1p, KB * 512, 64, 128, 128);

        // ---- SA2: npoint=128, ns=64, 131 -> 128,128,256 (Kpad 192) ----
        k_fps<<<dim3(KB), dim3(256), 0, stream>>>(l1x, 512, 128, fidx);
        k_gather<<<g1(KB * 128 * 3), dim3(256), 0, stream>>>(l1x, fidx, l2x, 512, 128, 3);
        k_ballquery<<<g1(KB * 128), dim3(256), 0, stream>>>(l1x, l2x, 512, 128, 64, r2 * r2, gidx);
        k_group<<<g1((size_t)KB * 128 * 64 * 192), dim3(256), 0, stream>>>(
            l1x, l2x, gidx, l1p, 512, 128, 64, 128, 192, ping);
        unsigned short* o2 = run_mlp(ping, KB * 128 * 64, 131, sa_ch[1], 3, saB1, nullptr, nullptr, 0);
        k_maxpool<<<g1(KB * 128 * 256), dim3(256), 0, stream>>>(o2, l2p, KB * 128, 64, 256, 256);

        // ---- SA3 (group_all): 259 -> 256,512,1024 (Kpad 320) ----
        k_groupall<<<g1((size_t)KB * 128 * 320), dim3(256), 0, stream>>>(l2x, l2p, 128, 256, 320, ping);
        unsigned short* o3 = run_mlp(ping, KB * 128, 259, sa_ch[2], 3, saB2, nullptr, nullptr, 0);
        k_maxpool<<<g1(KB * 1024), dim3(256), 0, stream>>>(o3, l3p, KB, 128, 1024, 1024);

        // ---- FP0: s==1 broadcast, 1280 -> 256,256 ----
        k_concat2<<<g1((size_t)KB * 128 * 1280), dim3(256), 0, stream>>>(
            l2p, 256, l3p, 1024, 128, 1, 1280, ping, (size_t)KB * 128);
        run_mlp(ping, KB * 128, 1280, fp_ch[0], 2, fpB0, l2pn, nullptr, 0);

        // ---- FP1: 3-NN interp, 384 -> 256,128 ----
        k_interp3<<<g1(KB * 512), dim3(256), 0, stream>>>(l1x, l2x, l2pn, interp, 512, 128, 256);
        k_concat2<<<g1((size_t)KB * 512 * 384), dim3(256), 0, stream>>>(
            l1p, 128, interp, 256, 0, 0, 384, ping, (size_t)KB * 512);
        run_mlp(ping, KB * 512, 384, fp_ch[1], 2, fpB1, l1pn, nullptr, 0);

        // ---- FP2: 3-NN interp, 134 -> 128,128,128 (Kpad 192) ----
        k_interp3<<<g1(KB * KN), dim3(256), 0, stream>>>(xyzp, l1x, l1pn, interp, KN, 512, 128);
        k_concat2<<<g1((size_t)KB * KN * 192), dim3(256), 0, stream>>>(
            l0cat, 6, interp, 128, 0, 0, 192, ping, (size_t)KB * KN);
        run_mlp(ping, KB * KN, 134, fp_ch[2], 3, fpB2, nullptr, l0pb[br], 128);
    }

    // ---- Head: 128->128 (x2, concat 256), 256->512, 512->128, then 128->32->1->1 ----
    unsigned short* featcat = ping;
    const float* W0 = (const float*)d_in[129]; const float* b0 = (const float*)d_in[130];
    const float* W1 = (const float*)d_in[131]; const float* b1 = (const float*)d_in[132];
    const float* W2 = (const float*)d_in[133]; const float* b2 = (const float*)d_in[134];
    const float* W3 = (const float*)d_in[135]; const float* b3 = (const float*)d_in[136];
    unsigned short* wb0 = prepw(W0, 128, 128);
    unsigned short* wb1 = prepw(W1, 128, 128);
    unsigned short* wb2 = prepw(W2, 512, 256);
    unsigned short* wb3 = prepw(W3, 128, 512);
    gemm(l0pb[0], KB * KN, 128, wb0, 128, 128, b0, nullptr, nullptr, 0, featcat, nullptr, 256, 0);
    gemm(l0pb[1], KB * KN, 128, wb1, 128, 128, b1, nullptr, nullptr, 0, featcat, nullptr, 256, 128);
    gemm(featcat, KB * KN, 256, wb2, 512, 512, b2, nullptr, nullptr, 0, pong, nullptr, 512, 0);
    gemm(pong,    KB * KN, 512, wb3, 128, 128, b3, nullptr, nullptr, 0, ping, nullptr, 128, 0);
    k_head_tail<<<dim3(KB * KN / 256), dim3(256), 0, stream>>>(
        ping,
        (const float*)d_in[137], (const float*)d_in[138],
        (const float*)d_in[139], (const float*)d_in[140],
        (const float*)d_in[141], (const float*)d_in[142],
        (float*)d_out, KB * KN);
}
